// DLRM_6176162971819
// MI455X (gfx1250) — compile-verified, matrix operand order checked by reading, not run on hardware
//
#include <hip/hip_runtime.h>
#include <hip/hip_bf16.h>
#include <math.h>

typedef __attribute__((ext_vector_type(16))) _Float16 v16h;
typedef __attribute__((ext_vector_type(8)))  _Float16 v8h;
typedef __attribute__((ext_vector_type(8)))  float    v8f;
typedef __attribute__((ext_vector_type(4)))  int      v4i;

#define BATCH  8192
#define EMBED  128
#define NCAT   26
#define NFEAT  27      // 1 + 26
#define NPAD   32
#define NTRIU  378     // 27*28/2 (upper triangle incl. diagonal)
#define TOPKP  512     // 128 + 378 = 506, padded to 512
#define TOPK   506
#define VOCAB  100000

// ---- gfx1250 async global->LDS copy (guarded; falls back to sync b128) ----
#ifdef __has_builtin
#  if __has_builtin(__builtin_amdgcn_global_load_async_to_lds_b128)
#    define HAVE_ASYNC_LDS 1
#  endif
#endif

#ifdef HAVE_ASYNC_LDS
#  if __has_builtin(__builtin_amdgcn_s_wait_asynccnt)
#    define WAIT_ASYNC() __builtin_amdgcn_s_wait_asynccnt(0)
#  else
#    define WAIT_ASYNC() asm volatile("s_wait_asynccnt 0" ::: "memory")
#  endif
#else
#  define WAIT_ASYNC() do {} while (0)
#endif

__device__ __forceinline__ void copy16B(const _Float16* g, _Float16* l) {
#ifdef HAVE_ASYNC_LDS
    typedef __attribute__((address_space(1))) v4i* gv4p;   // global int4*
    typedef __attribute__((address_space(3))) v4i* lv4p;   // LDS int4*
    v4i* gg = (v4i*)const_cast<_Float16*>(g);
    v4i* ll = (v4i*)l;
    __builtin_amdgcn_global_load_async_to_lds_b128((gv4p)gg, (lv4p)ll, 0, 0);
#else
    *(v8h*)l = *(const v8h*)g;
#endif
}

// Load a 16-bit WMMA A/B fragment from LDS ([row][32k] layout, 16B aligned).
// ISA layout: lanes 0-15 hold K=0..7,16..23 of row (lane&15); lanes 16-31 hold
// K=8..15,24..31. B stored as [N][K] uses the identical pattern (lane = col).
__device__ __forceinline__ v16h load_frag(const _Float16* lds, int rowStride,
                                          int row, int k0, int lane) {
    int koff = (lane & 16) ? 8 : 0;
    const _Float16* p = lds + (size_t)row * rowStride + k0 + koff;
    v8h lo = *(const v8h*)(p);
    v8h hi = *(const v8h*)(p + 16);
    return __builtin_shufflevector(lo, hi, 0, 1, 2, 3, 4, 5, 6, 7,
                                   8, 9, 10, 11, 12, 13, 14, 15);
}

// ---------------------------------------------------------------------------
// out[M,N](f16) = act(A[M,K](f16) @ Wt[N,K](f16)^T + bias(f32)).
// Requires M%128==0, N%128==0, K%32==0. Block 256 thr = 8 waves (4x2),
// 128x128 tile, each wave 32x64 (8 WMMA / K-step). Double-buffered LDS with
// async global->LDS copies.
// ---------------------------------------------------------------------------
#define TM 128
#define TN 128
#define TK 32

__global__ void __launch_bounds__(256)
gemm_f16(const _Float16* __restrict__ A, const _Float16* __restrict__ Wt,
         const float* __restrict__ bias, _Float16* __restrict__ out,
         int M, int K, int N, int relu) {
    __shared__ __attribute__((aligned(16))) _Float16 ldsA[2][TM * TK];
    __shared__ __attribute__((aligned(16))) _Float16 ldsB[2][TN * TK];

    const int tid  = threadIdx.x;
    const int lane = tid & 31;
    const int wave = tid >> 5;
    const int waveM = wave >> 1;   // 0..3
    const int waveN = wave & 1;    // 0..1
    const int r0 = blockIdx.y * TM;
    const int c0 = blockIdx.x * TN;
    const int kTiles = K >> 5;

    // 512 16B chunks per tile; 2 per thread; chunk = c*256+tid => coalesced.
    auto copy_tile = [&](int buf, int kbase) {
#pragma unroll
        for (int c = 0; c < 2; ++c) {
            int chunk = c * 256 + tid;
            int row = chunk >> 2;
            int koff = (chunk & 3) * 8;
            copy16B(A + (size_t)(r0 + row) * K + kbase + koff,
                    &ldsA[buf][row * TK + koff]);
        }
#pragma unroll
        for (int c = 0; c < 2; ++c) {
            int chunk = c * 256 + tid;
            int row = chunk >> 2;   // n-index
            int koff = (chunk & 3) * 8;
            copy16B(Wt + (size_t)(c0 + row) * K + kbase + koff,
                    &ldsB[buf][row * TK + koff]);
        }
    };

    copy_tile(0, 0);

    v8f acc[2][4] = {};
    for (int kt = 0; kt < kTiles; ++kt) {
        const int cur = kt & 1;
        WAIT_ASYNC();
        __syncthreads();
        if (kt + 1 < kTiles) copy_tile(cur ^ 1, (kt + 1) << 5);

        v16h aF[2], bF[4];
#pragma unroll
        for (int i = 0; i < 2; ++i)
            aF[i] = load_frag(ldsA[cur], TK, waveM * 32 + i * 16 + (lane & 15), 0, lane);
#pragma unroll
        for (int j = 0; j < 4; ++j)
            bF[j] = load_frag(ldsB[cur], TK, waveN * 64 + j * 16 + (lane & 15), 0, lane);
#pragma unroll
        for (int i = 0; i < 2; ++i)
#pragma unroll
            for (int j = 0; j < 4; ++j)
                acc[i][j] = __builtin_amdgcn_wmma_f32_16x16x32_f16(
                    false, aF[i], false, bF[j], (short)0, acc[i][j], false, false);
    }

    // C layout: vgpr r, lanes 0-15 -> (m=r, n=lane); lanes 16-31 -> (m=8+r).
    const int rsel = (lane & 16) ? 8 : 0;
    const int nloc = lane & 15;
#pragma unroll
    for (int i = 0; i < 2; ++i) {
#pragma unroll
        for (int j = 0; j < 4; ++j) {
            int gc = c0 + waveN * 64 + j * 16 + nloc;
            float bv = bias[gc];
#pragma unroll
            for (int r = 0; r < 8; ++r) {
                int gr = r0 + waveM * 32 + i * 16 + rsel + r;
                float v = acc[i][j][r] + bv;
                if (relu) v = fmaxf(v, 0.0f);
                out[(size_t)gr * N + gc] = (_Float16)v;
            }
        }
    }
}

// ---------------------------------------------------------------------------
// Weight prep: W[K][N] f32 -> Wt[N][Kpad] f16 (transposed, zero K-pad).
// ---------------------------------------------------------------------------
__global__ void __launch_bounds__(256)
convert_transpose(const float* __restrict__ W, _Float16* __restrict__ Wt,
                  int K, int N, int Kpad) {
    size_t idx = (size_t)blockIdx.x * 256 + threadIdx.x;
    if (idx >= (size_t)N * Kpad) return;
    int n = (int)(idx / Kpad), k = (int)(idx % Kpad);
    float v = (k < K) ? W[(size_t)k * N + n] : 0.0f;
    Wt[idx] = (_Float16)v;
}

// dense 8192x13 f32 -> 8192x32 f16 zero-padded
__global__ void __launch_bounds__(256)
convert_pad_dense(const float* __restrict__ D, _Float16* __restrict__ Dh) {
    int idx = blockIdx.x * 256 + threadIdx.x;   // over 8192*32
    int r = idx >> 5, k = idx & 31;
    Dh[idx] = (_Float16)((k < 13) ? D[r * 13 + k] : 0.0f);
}

// ---------------------------------------------------------------------------
// Embedding gather + feats@feats^T (27x27 in 32x32 pad) per sample via WMMA.
// bot is f16; top_in is f16 [8192][512] = [bot(128) | triu(378) | 0-pad(6)].
// Block 128 thr = 4 waves; one sample per wave.
// ---------------------------------------------------------------------------
__global__ void __launch_bounds__(128)
interact_kernel(const _Float16* __restrict__ bot, const int* __restrict__ cat,
                const float* __restrict__ table, _Float16* __restrict__ top_in) {
    __shared__ __attribute__((aligned(16))) _Float16 feats[4][NPAD * EMBED]; // 32KB

    const int lane = threadIdx.x & 31;
    const int wave = threadIdx.x >> 5;
    const int b    = blockIdx.x * 4 + wave;
    _Float16* F = feats[wave];

    // Row 0: bottom-MLP output (f16, 256B per row -> 8B per lane)
    ((uint2*)F)[lane] = ((const uint2*)(bot + (size_t)b * EMBED))[lane];
    // Rows 1..26: embedding rows (fp32 512B; one float4 per lane, cvt to f16)
#pragma unroll 2
    for (int j = 0; j < NCAT; ++j) {
        size_t row = (size_t)cat[(size_t)b * NCAT + j] + (size_t)j * VOCAB;
        float4 v = ((const float4*)(table + row * EMBED))[lane];
        _Float16* dst = F + (j + 1) * EMBED + lane * 4;
        dst[0] = (_Float16)v.x; dst[1] = (_Float16)v.y;
        dst[2] = (_Float16)v.z; dst[3] = (_Float16)v.w;
    }
    // Rows 27..31: zero pad
    for (int e = lane; e < (NPAD - NFEAT) * EMBED; e += 32)
        F[NFEAT * EMBED + e] = (_Float16)0.0f;
    __syncthreads();

    // 32x32 = feats(32x128) @ feats^T; B fragments == A fragments (symmetric).
    v8f acc[2][2] = {};
#pragma unroll
    for (int kt = 0; kt < 4; ++kt) {
        v16h aF[2];
#pragma unroll
        for (int i = 0; i < 2; ++i)
            aF[i] = load_frag(F, EMBED, i * 16 + (lane & 15), kt * 32, lane);
#pragma unroll
        for (int i = 0; i < 2; ++i)
#pragma unroll
            for (int j = 0; j < 2; ++j)
                acc[i][j] = __builtin_amdgcn_wmma_f32_16x16x32_f16(
                    false, aF[i], false, aF[j], (short)0, acc[i][j], false, false);
    }
    __syncthreads();

    // Spill 32x32 fp32 C over the (now dead) feats region, extract triu.
    float* C = (float*)F;
    const int rsel = (lane & 16) ? 8 : 0;
    const int nloc = lane & 15;
#pragma unroll
    for (int i = 0; i < 2; ++i)
#pragma unroll
        for (int j = 0; j < 2; ++j)
#pragma unroll
            for (int r = 0; r < 8; ++r)
                C[(i * 16 + rsel + r) * 32 + (j * 16 + nloc)] = acc[i][j][r];
    __syncthreads();

    _Float16* trow = top_in + (size_t)b * TOPKP;
    ((uint2*)trow)[lane] = ((const uint2*)(bot + (size_t)b * EMBED))[lane];
    for (int t = lane; t < NTRIU; t += 32) {
        int i = 0, rem = t;
        while (rem >= NFEAT - i) { rem -= NFEAT - i; ++i; }
        trow[EMBED + t] = (_Float16)C[i * 32 + (i + rem)];
    }
    if (lane < TOPKP - TOPK) trow[TOPK + lane] = (_Float16)0.0f;
}

// ---------------------------------------------------------------------------
// Final 256 -> 1 layer + sigmoid. One wave per batch row. X is f16.
// ---------------------------------------------------------------------------
__global__ void __launch_bounds__(256)
final_layer(const _Float16* __restrict__ X, const float* __restrict__ T4,
            const float* __restrict__ tb4, float* __restrict__ out) {
    const int lane = threadIdx.x & 31;
    const int wave = threadIdx.x >> 5;
    const int row  = blockIdx.x * 8 + wave;
    const _Float16* x = X + (size_t)row * 256;
    float s = 0.0f;
#pragma unroll
    for (int i = 0; i < 8; ++i) {
        int k = lane + i * 32;
        s += (float)x[k] * T4[k];
    }
#pragma unroll
    for (int off = 16; off > 0; off >>= 1)
        s += __shfl_down(s, off, 32);
    if (lane == 0) {
        float v = s + tb4[0];
        out[row] = 1.0f / (1.0f + expf(-v));
    }
}

extern "C" void kernel_launch(void* const* d_in, const int* in_sizes, int n_in,
                              void* d_out, int out_size, void* d_ws, size_t ws_size,
                              hipStream_t stream) {
    const float* dense = (const float*)d_in[0];
    const int*   cat   = (const int*)  d_in[1];
    const float* table = (const float*)d_in[2];
    const float* W0 = (const float*)d_in[3];  const float* b0 = (const float*)d_in[4];
    const float* W1 = (const float*)d_in[5];  const float* b1 = (const float*)d_in[6];
    const float* W2 = (const float*)d_in[7];  const float* b2 = (const float*)d_in[8];
    const float* T0 = (const float*)d_in[9];  const float* tb0 = (const float*)d_in[10];
    const float* T1 = (const float*)d_in[11]; const float* tb1 = (const float*)d_in[12];
    const float* T2 = (const float*)d_in[13]; const float* tb2 = (const float*)d_in[14];
    const float* T3 = (const float*)d_in[15]; const float* tb3 = (const float*)d_in[16];
    const float* T4 = (const float*)d_in[17]; const float* tb4 = (const float*)d_in[18];

    _Float16* h = (_Float16*)d_ws;
    size_t off = 0;
    _Float16* bufA   = h + off; off += (size_t)BATCH * 1024;
    _Float16* bufB   = h + off; off += (size_t)BATCH * 1024;
    _Float16* bot    = h + off; off += (size_t)BATCH * EMBED;
    _Float16* topi   = h + off; off += (size_t)BATCH * TOPKP;
    _Float16* denseh = h + off; off += (size_t)BATCH * 32;
    _Float16* W0t    = h + off; off += (size_t)512 * 32;
    _Float16* W1t    = h + off; off += (size_t)256 * 512;
    _Float16* W2t    = h + off; off += (size_t)128 * 256;
    _Float16* T0t    = h + off; off += (size_t)1024 * 512;
    _Float16* T1t    = h + off; off += (size_t)1024 * 1024;
    _Float16* T2t    = h + off; off += (size_t)512 * 1024;
    _Float16* T3t    = h + off; off += (size_t)256 * 512;

    auto cvt = [&](const float* W, _Float16* Wt, int K, int N, int Kpad) {
        int total = N * Kpad;
        convert_transpose<<<dim3((total + 255) / 256), dim3(256), 0, stream>>>(
            W, Wt, K, N, Kpad);
    };
    convert_pad_dense<<<dim3(BATCH * 32 / 256), dim3(256), 0, stream>>>(dense, denseh);
    cvt(W0, W0t, 13,   512,  32);
    cvt(W1, W1t, 512,  256,  512);
    cvt(W2, W2t, 256,  128,  256);
    cvt(T0, T0t, 506,  1024, 512);
    cvt(T1, T1t, 1024, 1024, 1024);
    cvt(T2, T2t, 1024, 512,  1024);
    cvt(T3, T3t, 512,  256,  512);

    auto gemm = [&](const _Float16* A, const _Float16* Wt, const float* bias,
                    _Float16* O, int M, int K, int N, int relu) {
        dim3 grid(N / TN, M / TM);
        gemm_f16<<<grid, dim3(256), 0, stream>>>(A, Wt, bias, O, M, K, N, relu);
    };

    // Bottom MLP: 13(->32) -> 512 -> 256 -> 128, ReLU
    gemm(denseh, W0t, b0, bufA, BATCH, 32,  512, 1);
    gemm(bufA,   W1t, b1, bufB, BATCH, 512, 256, 1);
    gemm(bufB,   W2t, b2, bot,  BATCH, 256, 128, 1);

    // Embedding gather + interactions -> top input (506, padded to 512)
    interact_kernel<<<dim3(BATCH / 4), dim3(128), 0, stream>>>(bot, cat, table, topi);

    // Top MLP: 512 -> 1024 -> 1024 -> 512 -> 256 (ReLU), then 256 -> 1 sigmoid
    gemm(topi, T0t, tb0, bufA, BATCH, 512,  1024, 1);
    gemm(bufA, T1t, tb1, bufB, BATCH, 1024, 1024, 1);
    gemm(bufB, T2t, tb2, bufA, BATCH, 1024, 512,  1);
    gemm(bufA, T3t, tb3, bufB, BATCH, 512,  256,  1);
    final_layer<<<dim3(BATCH / 8), dim3(256), 0, stream>>>(bufB, T4, tb4, (float*)d_out);
}
